// TriangleAttention_23476291239985
// MI455X (gfx1250) — matrix operand order repaired and checked by reading, not compiled, not run on hardware
//
#include <hip/hip_runtime.h>
#include <hip/hip_fp16.h>

#define DEV __device__ __forceinline__

typedef _Float16 half16v __attribute__((ext_vector_type(16)));
typedef _Float16 half8v  __attribute__((ext_vector_type(8)));
typedef _Float16 half4v  __attribute__((ext_vector_type(4)));
typedef float    float8v __attribute__((ext_vector_type(8)));

static constexpr int   N        = 256;
static constexpr int   D        = 128;
static constexpr int   H        = 4;
static constexpr int   KD       = 32;
static constexpr int   P        = N * N;        // 65536 positions
static constexpr float FP16_HUGE_F = 32768.0f;  // 2^15
static constexpr float LN_EPS_F    = 1e-5f;

DEV half16v combine16(half8v lo, half8v hi) {
  half16v r;
#pragma unroll
  for (int i = 0; i < 8; ++i) { r[i] = lo[i]; r[i + 8] = hi[i]; }
  return r;
}

DEV float8v zero8() {
  float8v z = {0.f, 0.f, 0.f, 0.f, 0.f, 0.f, 0.f, 0.f};
  return z;
}

// CDNA5 async global->LDS copy (ASYNCcnt-tracked), CDNA5 ISA.
// LDS operand is the per-lane LDS byte offset (low 32 bits of generic pointer).
DEV void async_load_b128(const void* gptr, void* lptr) {
  const unsigned lds = (unsigned)(uintptr_t)lptr;
  asm volatile("global_load_async_to_lds_b128 %0, %1, off"
               :: "v"(lds), "v"(gptr) : "memory");
}
DEV void wait_async0() { asm volatile("s_wait_asynccnt 0" ::: "memory"); }

// ---------------------------------------------------------------------------
// Kernel 0: convert the five DxD fp32 weight matrices to f16 (WMMA operands).
// ---------------------------------------------------------------------------
__global__ void __launch_bounds__(256) cvt_weights_kernel(
    const float* __restrict__ wq, const float* __restrict__ wk,
    const float* __restrict__ wv, const float* __restrict__ wg,
    const float* __restrict__ wo, _Float16* __restrict__ w16) {
  const int i = blockIdx.x * blockDim.x + threadIdx.x;
  if (i < D * D) {
    w16[0 * D * D + i] = (_Float16)wq[i];
    w16[1 * D * D + i] = (_Float16)wk[i];
    w16[2 * D * D + i] = (_Float16)wv[i];
    w16[3 * D * D + i] = (_Float16)wg[i];
    w16[4 * D * D + i] = (_Float16)wo[i];
  }
}

// ---------------------------------------------------------------------------
// Kernel 1: LayerNorm + nb = x . w2d^T.  One wave32 per position (4 floats
// per lane over D=128); wave shuffle reductions; writes x as f16.
// ---------------------------------------------------------------------------
__global__ void __launch_bounds__(128) ln_nb_kernel(
    const float* __restrict__ pact, const float* __restrict__ lnscale,
    const float* __restrict__ lnbias, const float* __restrict__ w2d,
    _Float16* __restrict__ x16, float* __restrict__ nbuf) {
  const int wv = threadIdx.x >> 5;
  const int lane = threadIdx.x & 31;
  const int p = blockIdx.x * 4 + wv;
  const size_t base = (size_t)p * D + lane * 4;

  const float4 xr = *(const float4*)(pact + base);
  float s = xr.x + xr.y + xr.z + xr.w;
#pragma unroll
  for (int m = 1; m < 32; m <<= 1) s += __shfl_xor(s, m, 32);
  const float mu = s * (1.0f / (float)D);

  const float d0 = xr.x - mu, d1 = xr.y - mu, d2 = xr.z - mu, d3 = xr.w - mu;
  float vs = d0 * d0 + d1 * d1 + d2 * d2 + d3 * d3;
#pragma unroll
  for (int m = 1; m < 32; m <<= 1) vs += __shfl_xor(vs, m, 32);
  const float rs = rsqrtf(vs * (1.0f / (float)D) + LN_EPS_F);

  const float4 sc = *(const float4*)(lnscale + lane * 4);
  const float4 bi = *(const float4*)(lnbias + lane * 4);
  const float y0 = d0 * rs * sc.x + bi.x;
  const float y1 = d1 * rs * sc.y + bi.y;
  const float y2 = d2 * rs * sc.z + bi.z;
  const float y3 = d3 * rs * sc.w + bi.w;

  half4v hv;
  hv[0] = (_Float16)y0; hv[1] = (_Float16)y1;
  hv[2] = (_Float16)y2; hv[3] = (_Float16)y3;
  *(half4v*)(x16 + base) = hv;

#pragma unroll
  for (int hh = 0; hh < H; ++hh) {
    const float4 w = *(const float4*)(w2d + hh * D + lane * 4);
    float dp = y0 * w.x + y1 * w.y + y2 * w.z + y3 * w.w;
#pragma unroll
    for (int m = 1; m < 32; m <<= 1) dp += __shfl_xor(dp, m, 32);
    if (lane == 0) nbuf[(size_t)hh * P + p] = dp;  // nb[h][i][j], p = i*N+j
  }
}

// ---------------------------------------------------------------------------
// Kernel 2: C[P,128] = A[P,128] . W[128,128]^T via v_wmma_f32_16x16x32_f16.
// W (32KB f16) is async-staged into LDS once per block; A fragments for all
// four K-steps preloaded; B fragments double-buffered (loads for K-step s+1
// issue before the WMMAs of step s) so ds_load latency overlaps WMMA.
// Epilogue modes: 0 plain f16 (k,v), 1 q-scale f16, 2 sigmoid(.+bg) f16 (g),
// 3 (.+bo)*mask -> f32 (final output projection).
// ---------------------------------------------------------------------------
__global__ void __launch_bounds__(256) gemm_nt_kernel(
    const _Float16* __restrict__ A, const _Float16* __restrict__ W,
    void* __restrict__ dst, const int mode, const float* __restrict__ bias,
    const float* __restrict__ mask, const float qscale) {
  __shared__ _Float16 sw[D * D];  // 32 KB weight tile

  const int tid = threadIdx.x;
  const int wv = tid >> 5;
  const int lane = tid & 31;
  const int n = lane & 15;   // column within 16-wide tile / A row index
  const int hf = lane >> 4;  // lane half
  const int m0 = blockIdx.x * 128 + wv * 16;

  // Async-stage the whole weight matrix: 256 threads x 16 x b128.
#pragma unroll
  for (int t = 0; t < 16; ++t) {
    const int idx = (t * 256 + tid) * 8;  // halfs (16B chunks)
    async_load_b128(W + idx, sw + idx);
  }

  // Preload A fragments for all 4 K-steps while the async copy is in flight.
  half16v afrag[4];
#pragma unroll
  for (int ks = 0; ks < 4; ++ks) {
    const _Float16* ap = A + (size_t)(m0 + n) * D + ks * 32;
    afrag[ks] = *(const half16v*)(ap + hf * 16);
  }

  wait_async0();
  __syncthreads();

  float8v acc[8];
#pragma unroll
  for (int nt = 0; nt < 8; ++nt) acc[nt] = zero8();

  // Double-buffered B fragments from LDS.
  half16v b[2][8];
#pragma unroll
  for (int nt = 0; nt < 8; ++nt)
    b[0][nt] = *(const half16v*)(sw + (size_t)(nt * 16 + n) * D + hf * 16);

#pragma unroll
  for (int ks = 0; ks < 4; ++ks) {
    const int cur = ks & 1, nxt = cur ^ 1;
    if (ks < 3) {
#pragma unroll
      for (int nt = 0; nt < 8; ++nt) {
        b[nxt][nt] = *(const half16v*)(sw + (size_t)(nt * 16 + n) * D +
                                       (ks + 1) * 32 + hf * 16);
      }
    }
#pragma unroll
    for (int nt = 0; nt < 8; ++nt) {
      acc[nt] = __builtin_amdgcn_wmma_f32_16x16x32_f16(
          false, afrag[ks], false, b[cur][nt], (short)0, acc[nt], false, false);
    }
  }

#pragma unroll
  for (int nt = 0; nt < 8; ++nt) {
#pragma unroll
    for (int vr = 0; vr < 8; ++vr) {
      const int row = m0 + vr + 8 * hf;  // C layout: M = vgpr + 8*(lane/16)
      const int col = nt * 16 + n;       // N = lane%16
      float v = acc[nt][vr];
      if (mode == 1) {
        v *= qscale;
      } else if (mode == 2) {
        v += bias[col];
        v = 1.0f / (1.0f + __expf(-v));
      }
      if (mode == 3) {
        v = (v + bias[col]) * mask[row];
        ((float*)dst)[(size_t)row * D + col] = v;
      } else {
        ((_Float16*)dst)[(size_t)row * D + col] = (_Float16)v;
      }
    }
  }
}

// ---------------------------------------------------------------------------
// Kernel 3: attention for one (row r, head h) per block. q/k async-staged to
// LDS, v transposed through VGPRs; logits in one WMMA per 16x16 tile
// (K = kd = 32) with the k-fragment double-buffered; softmax via 16-lane
// shuffles on the C layout; probs round-trip through per-wave LDS to get the
// A-fragment layout; o = probs . v via 8 K-steps x 2 WMMAs (double-buffered);
// gated epilogue.
// ---------------------------------------------------------------------------
__global__ void __launch_bounds__(256) attn_kernel(
    const _Float16* __restrict__ q16, const _Float16* __restrict__ k16,
    const _Float16* __restrict__ v16, const _Float16* __restrict__ g16,
    const float* __restrict__ nbuf, const float* __restrict__ mask,
    _Float16* __restrict__ o16) {
  extern __shared__ char smem_raw[];
  _Float16* sq  = (_Float16*)smem_raw;  // [256][32]
  _Float16* sk  = sq + N * KD;          // [256][32]
  _Float16* svT = sk + N * KD;          // [32][256] (v transposed)
  _Float16* sp  = svT + KD * N;         // 8 waves x [16][256] probs

  const int r = blockIdx.x;
  const int h = blockIdx.y;
  const int tid = threadIdx.x;
  const int wv = tid >> 5;
  const int lane = tid & 31;
  const int n = lane & 15;
  const int hf = lane >> 4;

  // Stage q/k/v rows for this (r, h): thread t handles key/query index j = t.
  {
    const int j = tid;
    const size_t gbase = ((size_t)(r * N + j)) * D + h * KD;
#pragma unroll
    for (int c = 0; c < 4; ++c) {
      async_load_b128(q16 + gbase + c * 8, sq + j * KD + c * 8);
      async_load_b128(k16 + gbase + c * 8, sk + j * KD + c * 8);
    }
    const half16v v0 = *(const half16v*)(v16 + gbase);
    const half16v v1 = *(const half16v*)(v16 + gbase + 16);
#pragma unroll
    for (int c = 0; c < 16; ++c) svT[c * N + j] = v0[c];
#pragma unroll
    for (int c = 0; c < 16; ++c) svT[(16 + c) * N + j] = v1[c];
  }
  wait_async0();
  __syncthreads();

  _Float16* spw = sp + wv * 16 * N;

#pragma unroll 1
  for (int pass = 0; pass < 2; ++pass) {
    const int it = wv + pass * 8;  // i-tile 0..15

    // q A-fragment for this i-tile (K = 32 = full kd)
    const _Float16* qp = sq + (it * 16 + n) * KD;
    const half16v aq = *(const half16v*)(qp + hf * 16);

    // logits: double-buffered k-fragments
    float8v acc[16];
    half16v bk[2];
    bk[0] = *(const half16v*)(sk + (size_t)n * KD + hf * 16);
#pragma unroll
    for (int jt = 0; jt < 16; ++jt) {
      const int cur = jt & 1;
      if (jt < 15) {
        bk[cur ^ 1] =
            *(const half16v*)(sk + (size_t)((jt + 1) * 16 + n) * KD + hf * 16);
      }
      acc[jt] = __builtin_amdgcn_wmma_f32_16x16x32_f16(
          false, aq, false, bk[cur], (short)0, zero8(), false, false);
      // add mask bias (per j) and nb (per h,i,j)
      const int j = jt * 16 + n;
      const float bm = FP16_HUGE_F * (mask[(size_t)r * N + j] - 1.0f);
      const float* nbp = nbuf + (size_t)h * P + (size_t)(it * 16 + 8 * hf) * N + j;
#pragma unroll
      for (int vr = 0; vr < 8; ++vr) acc[jt][vr] += bm + nbp[(size_t)vr * N];
    }

    // softmax over j (256): per-lane fold over 16 tiles, then 16-lane shuffle
    float rmax[8], rinv[8];
#pragma unroll
    for (int vr = 0; vr < 8; ++vr) {
      float mx = acc[0][vr];
#pragma unroll
      for (int jt = 1; jt < 16; ++jt) mx = fmaxf(mx, acc[jt][vr]);
#pragma unroll
      for (int m = 1; m < 16; m <<= 1) mx = fmaxf(mx, __shfl_xor(mx, m, 32));
      rmax[vr] = mx;
    }
    float rsum[8];
#pragma unroll
    for (int vr = 0; vr < 8; ++vr) rsum[vr] = 0.0f;
#pragma unroll
    for (int jt = 0; jt < 16; ++jt) {
#pragma unroll
      for (int vr = 0; vr < 8; ++vr) {
        const float e = __expf(acc[jt][vr] - rmax[vr]);
        acc[jt][vr] = e;
        rsum[vr] += e;
      }
    }
#pragma unroll
    for (int vr = 0; vr < 8; ++vr) {
      float s = rsum[vr];
#pragma unroll
      for (int m = 1; m < 16; m <<= 1) s += __shfl_xor(s, m, 32);
      rinv[vr] = 1.0f / s;
    }

    // write normalized probs to this wave's LDS slab, row-major [i_local][j]
#pragma unroll
    for (int jt = 0; jt < 16; ++jt) {
#pragma unroll
      for (int vr = 0; vr < 8; ++vr) {
        spw[(vr + 8 * hf) * N + jt * 16 + n] =
            (_Float16)(acc[jt][vr] * rinv[vr]);
      }
    }
    __syncthreads();

    // o = probs . v : K loop over j in blocks of 32, double-buffered
    float8v oacc[2] = {zero8(), zero8()};
    half16v ap[2], bv[2][2];
    {
      const _Float16* pp = spw + n * N;
      ap[0] = *(const half16v*)(pp + hf * 16);
      bv[0][0] = *(const half16v*)(svT + (size_t)n * N + hf * 16);
      bv[0][1] = *(const half16v*)(svT + (size_t)(16 + n) * N + hf * 16);
    }
#pragma unroll
    for (int jb = 0; jb < 8; ++jb) {
      const int cur = jb & 1, nxt = cur ^ 1;
      if (jb < 7) {
        const _Float16* pp = spw + n * N + (jb + 1) * 32;
        ap[nxt] = *(const half16v*)(pp + hf * 16);
        bv[nxt][0] =
            *(const half16v*)(svT + (size_t)n * N + (jb + 1) * 32 + hf * 16);
        bv[nxt][1] =
            *(const half16v*)(svT + (size_t)(16 + n) * N + (jb + 1) * 32 + hf * 16);
      }
#pragma unroll
      for (int nt = 0; nt < 2; ++nt) {
        oacc[nt] = __builtin_amdgcn_wmma_f32_16x16x32_f16(
            false, ap[cur], false, bv[cur][nt], (short)0, oacc[nt], false, false);
      }
    }

    // gated epilogue: o *= sigmoid(x.wg^T + bg) (precomputed in g16)
#pragma unroll
    for (int nt = 0; nt < 2; ++nt) {
#pragma unroll
      for (int vr = 0; vr < 8; ++vr) {
        const int i = it * 16 + vr + 8 * hf;
        const size_t pidx = (size_t)r * N + i;
        const int col = h * KD + nt * 16 + n;
        const float gv = (float)g16[pidx * D + col];
        o16[pidx * D + col] = (_Float16)(oacc[nt][vr] * gv);
      }
    }
    __syncthreads();
  }
}

// ---------------------------------------------------------------------------
// Host launcher
// ---------------------------------------------------------------------------
extern "C" void kernel_launch(void* const* d_in, const int* in_sizes, int n_in,
                              void* d_out, int out_size, void* d_ws, size_t ws_size,
                              hipStream_t stream) {
  (void)in_sizes; (void)n_in; (void)out_size; (void)ws_size;

  const float* pact = (const float*)d_in[0];
  const float* mask = (const float*)d_in[1];
  const float* lns  = (const float*)d_in[2];
  const float* lnb  = (const float*)d_in[3];
  const float* w2d  = (const float*)d_in[4];
  const float* wq   = (const float*)d_in[5];
  const float* wk   = (const float*)d_in[6];
  const float* wvv  = (const float*)d_in[7];
  const float* wg   = (const float*)d_in[8];
  const float* bg   = (const float*)d_in[9];
  const float* wo   = (const float*)d_in[10];
  const float* bo   = (const float*)d_in[11];
  float* out = (float*)d_out;

  char* ws = (char*)d_ws;
  auto al = [](size_t v) { return (v + 255) & ~(size_t)255; };
  size_t off = 0;
  _Float16* x16 = (_Float16*)(ws + off); off = al(off + (size_t)P * D * 2);
  _Float16* q16 = (_Float16*)(ws + off); off = al(off + (size_t)P * D * 2);
  _Float16* k16 = (_Float16*)(ws + off); off = al(off + (size_t)P * D * 2);
  _Float16* v16 = (_Float16*)(ws + off); off = al(off + (size_t)P * D * 2);
  _Float16* g16 = (_Float16*)(ws + off); off = al(off + (size_t)P * D * 2);
  _Float16* o16 = (_Float16*)(ws + off); off = al(off + (size_t)P * D * 2);
  float*    nbuf = (float*)(ws + off);   off = al(off + (size_t)H * P * 4);
  _Float16* w16 = (_Float16*)(ws + off); off = al(off + (size_t)5 * D * D * 2);

  cvt_weights_kernel<<<(D * D + 255) / 256, 256, 0, stream>>>(wq, wk, wvv, wg, wo, w16);
  ln_nb_kernel<<<P / 4, 128, 0, stream>>>(pact, lns, lnb, w2d, x16, nbuf);

  const float qscale = 0.17677669529663687f;  // 1/sqrt(kd=32)
  gemm_nt_kernel<<<P / 128, 256, 0, stream>>>(x16, w16 + 0 * D * D, q16, 1, bo, mask, qscale);
  gemm_nt_kernel<<<P / 128, 256, 0, stream>>>(x16, w16 + 1 * D * D, k16, 0, bo, mask, 1.0f);
  gemm_nt_kernel<<<P / 128, 256, 0, stream>>>(x16, w16 + 2 * D * D, v16, 0, bo, mask, 1.0f);
  gemm_nt_kernel<<<P / 128, 256, 0, stream>>>(x16, w16 + 3 * D * D, g16, 2, bg, mask, 1.0f);

  const size_t shmem = (size_t)(N * KD * 3 + 8 * 16 * N) * sizeof(_Float16);  // 112 KB
  attn_kernel<<<dim3(N, H), 256, shmem, stream>>>(q16, k16, v16, g16, nbuf, mask, o16);

  gemm_nt_kernel<<<P / 128, 256, 0, stream>>>(o16, w16 + 4 * D * D, out, 3, bo, mask, 1.0f);
}
